// SampledKernelXML_51986284151005
// MI455X (gfx1250) — compile-verified, matrix operand order checked by reading, not run on hardware
//
#include <hip/hip_runtime.h>
#include <hip/hip_bf16.h>
#include <math.h>

// ---- problem constants (from reference) ----
#define BB    1024
#define LL    128
#define KK    8
#define DD    256
#define HH    4
#define HDD   64
#define PP    16
#define MM    32
#define RR    2
#define FF    4096     // R*H*P*M
#define NNEG  2048
#define NPAIR 8192     // B*K
#define NLAB  131072

// Gauss-Laguerre(2) derived constants (nodes/weights / C, C = 2.000001)
__device__ __constant__ float c_s[2]     = {0.29289307f, 1.70710593f};   // s = max(node,.01)/C
__device__ __constant__ float c_sq2s[2]  = {0.76536686f, 1.84775907f};   // sqrt(2s)
__device__ __constant__ float c_scale[2] = {0.11548494f, 0.04783543f};   // sqrt(w/C)/sqrt(M+1e-6)

typedef __attribute__((ext_vector_type(16))) __bf16 v16bf;
typedef __attribute__((ext_vector_type(8)))  float  v8f;

// ---- workspace layout (floats) ----
#define OFF_ANC    0
#define OFF_QUERY  (OFF_ANC   + PP*HDD)
#define OFF_QPOLY  (OFF_QUERY + BB*DD)
#define OFF_QPRF   (OFF_QPOLY + BB*HH*PP)
#define OFF_WPOLY  (OFF_QPRF  + BB*RR*HH*MM)
#define OFF_WPRF   (OFF_WPOLY + NPAIR*HH*PP)
#define OFF_S      (OFF_WPRF  + NPAIR*RR*HH*MM)
#define OFF_POS    (OFF_S     + FF)
#define OFF_NEGD   (OFF_POS   + NPAIR)

// ---------------- anchors / ||anchors|| ----------------
__global__ void k_norm_anchors(const float* __restrict__ anchors, float* __restrict__ anc) {
  __shared__ float red[HDD];
  int p = blockIdx.x, t = threadIdx.x;
  float v = anchors[p * HDD + t];
  red[t] = v * v; __syncthreads();
  for (int s = HDD / 2; s > 0; s >>= 1) { if (t < s) red[t] += red[t + s]; __syncthreads(); }
  anc[p * HDD + t] = v / fmaxf(sqrtf(red[0]), 1e-12f);
}

// ---------------- masked-mean query (HBM-bound stage; prefetch next row) ----------------
__global__ void k_query(const int* __restrict__ idx, const float* __restrict__ mask,
                        const float* __restrict__ emb, float* __restrict__ query) {
  int b = blockIdx.x, d = threadIdx.x;
  float acc = 0.f, ms = 0.f;
  for (int l = 0; l < LL; ++l) {
    int ix = idx[b * LL + l];
    if (l + 1 < LL) {
      int ixn = idx[b * LL + l + 1];
      __builtin_prefetch(&emb[(size_t)ixn * DD + d], 0, 0);   // global_prefetch_b8
    }
    float mk = mask[b * LL + l];
    acc += emb[(size_t)ix * DD + d] * mk;
    ms  += mk;
  }
  // global safe_normalize absorbed by the per-head normalize in k_feats
  query[b * DD + d] = acc / fmaxf(ms, 1.0f);
}

// ---------------- SLAY factor kernel ----------------
// mode 0: query rows -> poly/prf out
// mode 1: pos label rows (gather kernel column) -> poly/prf out
// mode 2: neg rows -> accumulate full phi into Ssum (4096) with atomics
__global__ void k_feats(int mode,
                        const float* __restrict__ query, const int* __restrict__ labels,
                        const int* __restrict__ negidx,  const float* __restrict__ Wk,
                        const float* __restrict__ anc,   const float* __restrict__ omega,
                        float* __restrict__ polyOut, float* __restrict__ prfOut,
                        float* __restrict__ Ssum) {
  __shared__ float x[DD];
  __shared__ float inv[HH];
  __shared__ float poly[HH * PP];
  __shared__ float prf[RR * HH * MM];
  int row = blockIdx.x, t = threadIdx.x;

  if (mode == 0) {
    x[t] = query[row * DD + t];
  } else {
    int lbl = (mode == 1) ? max(labels[row], 0) : negidx[row];
    x[t] = Wk[(size_t)t * NLAB + lbl];           // kernel.T row gather
  }
  __syncthreads();

  if (t < HH) {                                   // per-head inverse norm
    float s = 0.f;
    for (int d = 0; d < HDD; ++d) { float v = x[t * HDD + d]; s += v * v; }
    inv[t] = 1.0f / fmaxf(sqrtf(s), 1e-4f);
  }
  __syncthreads();

  if (t < HH * PP) {                              // poly_feat = clip(xr.anc)^2 / sqrt(P)
    int h = t >> 4, p = t & 15;
    float a = 0.f;
    for (int d = 0; d < HDD; ++d) a += x[h * HDD + d] * anc[p * HDD + d];
    a *= inv[h];
    a = fminf(fmaxf(a, -1.f), 1.f);
    poly[t] = a * a * 0.25f;
  }
  {                                               // prf_feat (includes quad weights)
    int r = t >> 7, h = (t >> 5) & 3, m = t & 31;
    const float* om = omega + (size_t)((r * HH + h) * HDD) * MM + m;
    float a = 0.f;
    for (int d = 0; d < HDD; ++d) a += x[h * HDD + d] * om[d * MM];
    a *= inv[h] * 0.125f;                         // omega / sqrt(HD)
    float arg = fminf(fmaxf(a * c_sq2s[r] - c_s[r], -20.f), 20.f);
    prf[t] = __expf(arg) * c_scale[r];
  }
  __syncthreads();

  if (mode == 2) {
    #pragma unroll
    for (int j = 0; j < FF / DD; ++j) {
      int f = t + DD * j;
      int m = f & 31, p = (f >> 5) & 15, h = (f >> 9) & 3, r = (f >> 11) & 1;
      atomicAdd(&Ssum[f], poly[h * PP + p] * prf[r * HH * MM + h * MM + m]);
    }
  } else {
    if (t < HH * PP) polyOut[row * HH * PP + t] = poly[t];
    prfOut[row * RR * HH * MM + t] = prf[t];
  }
}

// one 512B async panel copy: 32 lanes x b128, VGPR-bypassing, ASYNCcnt-tracked
__device__ __forceinline__ void async_copy512(unsigned lds_base, const float* gbase,
                                              unsigned gbyte_off, int lane) {
  unsigned la = lds_base + (unsigned)lane * 16u;
  unsigned va = gbyte_off + (unsigned)lane * 16u;
  asm volatile("global_load_async_to_lds_b128 %0, %1, %2"
               :: "v"(la), "v"(va), "s"(gbase) : "memory");
}

// ---------------- pos scores: 16-pair tiles via v_wmma_f32_16x16x32_bf16 ----------------
// A tile touches only 2 query rows (pair g=blk*16+i -> b=g>>3), so all four factor
// panels are contiguous -> async b128 staging. Fragments rebuilt on the fly from
// poly/prf per the bf16 A-layout (lane&15=row, VGPR0-3 -> K 0..7|8..15, VGPR4-7 -> +16).
__global__ void k_pos_wmma(const float* __restrict__ qpoly, const float* __restrict__ qprf,
                           const float* __restrict__ wpoly, const float* __restrict__ wprf,
                           float* __restrict__ pos) {
  __shared__ float aPoly[2 * 64];
  __shared__ float aPrf [2 * 256];
  __shared__ float bPoly[16 * 64];
  __shared__ float bPrf [16 * 256];
  __shared__ float cdiag[16];
  int blk = blockIdx.x, lane = threadIdx.x;

  {
    unsigned aPolyB = (unsigned)(size_t)(float*)aPoly;
    unsigned aPrfB  = (unsigned)(size_t)(float*)aPrf;
    unsigned bPolyB = (unsigned)(size_t)(float*)bPoly;
    unsigned bPrfB  = (unsigned)(size_t)(float*)bPrf;
    async_copy512(aPolyB, qpoly, (unsigned)blk * 512u, lane);                 // 2x64 f32
    #pragma unroll
    for (int j = 0; j < 4; ++j)                                              // 2x256 f32
      async_copy512(aPrfB + j * 512u, qprf, (unsigned)blk * 2048u + j * 512u, lane);
    #pragma unroll
    for (int j = 0; j < 8; ++j)                                              // 16x64 f32
      async_copy512(bPolyB + j * 512u, wpoly, (unsigned)blk * 4096u + j * 512u, lane);
    #pragma unroll
    for (int j = 0; j < 32; ++j)                                             // 16x256 f32
      async_copy512(bPrfB + j * 512u, wprf, (unsigned)blk * 16384u + j * 512u, lane);
    asm volatile("s_wait_asynccnt 0" ::: "memory");
  }
  __syncthreads();

  int row  = lane & 15;       // tile row / col held by this lane
  int arow = row >> 3;        // which of the 2 query rows
  int kb   = (lane & 16) ? 8 : 0;
  v8f c = {};
  for (int kc = 0; kc < FF; kc += 64) {
    #pragma unroll
    for (int half = 0; half < 2; ++half) {
      int kbase = kc + half * 32;
      v16bf av, bv;
      #pragma unroll
      for (int e = 0; e < 16; ++e) {
        int v  = e >> 1;
        int Kk = ((v < 4) ? (kb + 2 * v) : (16 + kb + 2 * (v - 4))) + (e & 1);
        int f  = kbase + Kk;
        int m = f & 31, p = (f >> 5) & 15, h = (f >> 9) & 3, r = (f >> 11) & 1;
        int pi = h * 16 + p, fi = r * 128 + h * 32 + m;
        av[e] = (__bf16)(aPoly[arow * 64 + pi] * aPrf[arow * 256 + fi]);
        bv[e] = (__bf16)(bPoly[row  * 64 + pi] * bPrf[row  * 256 + fi]);
      }
      c = __builtin_amdgcn_wmma_f32_16x16x32_bf16(false, av, false, bv, (short)0, c, false, false);
    }
  }
  // diagonal of 16x16 f32 C: VGPR vr holds (M=vr, N=lane) / (M=vr+8, N=lane-16)
  #pragma unroll
  for (int vr = 0; vr < 8; ++vr) {
    int Mrow = (lane & 16) ? (vr + 8) : vr;
    if ((lane & 15) == Mrow) cdiag[Mrow] = c[vr];
  }
  __syncthreads();
  if (lane < 16) pos[blk * 16 + lane] = cdiag[lane];
}

// ---------------- neg contribution: phi_q[b] . sum_neg_phi ----------------
__global__ void k_negdot(const float* __restrict__ qpoly, const float* __restrict__ qprf,
                         const float* __restrict__ S, float* __restrict__ negd) {
  __shared__ float sp[64], sf[256], red[256];
  int b = blockIdx.x, t = threadIdx.x;
  if (t < 64) sp[t] = qpoly[b * 64 + t];
  sf[t] = qprf[b * 256 + t];
  __syncthreads();
  float acc = 0.f;
  #pragma unroll
  for (int j = 0; j < FF / DD; ++j) {
    int f = t + DD * j;
    int m = f & 31, p = (f >> 5) & 15, h = (f >> 9) & 3, r = (f >> 11) & 1;
    acc += sp[h * 16 + p] * sf[r * 128 + h * 32 + m] * S[f];
  }
  red[t] = acc; __syncthreads();
  for (int s = 128; s > 0; s >>= 1) { if (t < s) red[t] += red[t + s]; __syncthreads(); }
  if (t == 0) negd[b] = red[0];
}

// ---------------- loss reduction ----------------
__global__ void k_loss(const float* __restrict__ pos, const float* __restrict__ negd,
                       const float* __restrict__ lm, float* __restrict__ out) {
  __shared__ float rn[256], rd[256];
  int t = threadIdx.x;
  float num = 0.f, den = 0.f;
  for (int b = t; b < BB; b += 256) {
    float Z = negd[b] + 2048.0f * 1e-8f;          // + eps per neg column
    float pk[KK];
    #pragma unroll
    for (int k = 0; k < KK; ++k) { pk[k] = fmaxf(pos[b * KK + k] + 1e-8f, 1e-8f); Z += pk[k]; }
    float logZ = __logf(fmaxf(Z, 1e-8f));         // logsumexp(log(s)) == log(sum(s))
    #pragma unroll
    for (int k = 0; k < KK; ++k) {
      float l = lm[b * KK + k];
      num += (__logf(pk[k]) - logZ) * l;
      den += l;
    }
  }
  rn[t] = num; rd[t] = den; __syncthreads();
  for (int s = 128; s > 0; s >>= 1) { if (t < s) { rn[t] += rn[t + s]; rd[t] += rd[t + s]; } __syncthreads(); }
  if (t == 0) out[0] = -rn[0] / (rd[0] + 1e-6f);
}

extern "C" void kernel_launch(void* const* d_in, const int* in_sizes, int n_in,
                              void* d_out, int out_size, void* d_ws, size_t ws_size,
                              hipStream_t stream) {
  const int*   indices    = (const int*)  d_in[0];
  const float* mask       = (const float*)d_in[1];
  const int*   labels     = (const int*)  d_in[2];
  const float* label_mask = (const float*)d_in[3];
  const int*   negidx     = (const int*)  d_in[4];
  const float* emb        = (const float*)d_in[5];
  const float* Wk         = (const float*)d_in[6];
  const float* omega      = (const float*)d_in[7];
  const float* anchors    = (const float*)d_in[8];

  float* ws    = (float*)d_ws;
  float* anc   = ws + OFF_ANC;
  float* query = ws + OFF_QUERY;
  float* qpoly = ws + OFF_QPOLY;
  float* qprf  = ws + OFF_QPRF;
  float* wpoly = ws + OFF_WPOLY;
  float* wprf  = ws + OFF_WPRF;
  float* S     = ws + OFF_S;
  float* pos   = ws + OFF_POS;
  float* negd  = ws + OFF_NEGD;

  hipMemsetAsync(S, 0, FF * sizeof(float), stream);

  k_norm_anchors<<<PP, HDD, 0, stream>>>(anchors, anc);
  k_query<<<BB, DD, 0, stream>>>(indices, mask, emb, query);

  // SLAY factors: queries, positive label rows, negative rows (accumulate sum-phi)
  k_feats<<<BB,    DD, 0, stream>>>(0, query, nullptr, nullptr, Wk, anc, omega, qpoly, qprf, nullptr);
  k_feats<<<NPAIR, DD, 0, stream>>>(1, nullptr, labels, nullptr, Wk, anc, omega, wpoly, wprf, nullptr);
  k_feats<<<NNEG,  DD, 0, stream>>>(2, nullptr, nullptr, negidx, Wk, anc, omega, nullptr, nullptr, S);

  k_pos_wmma<<<NPAIR / 16, 32, 0, stream>>>(qpoly, qprf, wpoly, wprf, pos);
  k_negdot<<<BB, 256, 0, stream>>>(qpoly, qprf, S, negd);
  k_loss<<<1, 256, 0, stream>>>(pos, negd, label_mask, (float*)d_out);
}
